// VPGNet_31464930411245
// MI455X (gfx1250) — matrix operand order repaired and checked by reading, not compiled, not run on hardware
//
#include <hip/hip_runtime.h>
#include <hip/hip_bf16.h>
#include <math.h>

typedef _Float16 half_t;
typedef __attribute__((ext_vector_type(16))) _Float16 v16h;
typedef __attribute__((ext_vector_type(8)))  _Float16 v8h;
typedef __attribute__((ext_vector_type(4)))  _Float16 v4h;
typedef __attribute__((ext_vector_type(8)))  float    v8f;
typedef __attribute__((ext_vector_type(4)))  float    v4f;

// ---------------------------------------------------------------------------
// WMMA-B fragment-ordered panel index.
// Panel stores B[k][m] (k < K, m < Mpad) blocked so that one wave's B operand
// for a 32-k block and 16-m block is 32 lanes x 16 contiguous halfs:
//   lane = (k%32)/16 * 16 + (m%16), slot j = k%16
// ---------------------------------------------------------------------------
__device__ __forceinline__ int bfrag_idx(int k, int m, int Mp16)
{
    int kb   = k >> 5;
    int lane = (((k >> 4) & 1) << 4) | (m & 15);
    int j    = k & 15;
    int mb   = m >> 4;
    return ((kb * Mp16 + mb) * 32 + lane) * 16 + j;
}

// ---------------------------------------------------------------------------
// Generic direct convolution (fp32), one thread per output element.
// Used for conv1..conv5 (awkward shapes / small share of total FLOPs).
// ---------------------------------------------------------------------------
__global__ void vpg_conv2d_kernel(const float* __restrict__ in,
                                  const float* __restrict__ w,
                                  const float* __restrict__ bias,
                                  float* __restrict__ out,
                                  int IC, int IH, int IW,
                                  int OC, int OH, int OW,
                                  int KH, int KW, int stride, int pad, int relu)
{
    int idx = blockIdx.x * blockDim.x + threadIdx.x;
    int total = OC * OH * OW;
    if (idx >= total) return;
    int ow = idx % OW;
    int oh = (idx / OW) % OH;
    int oc = idx / (OW * OH);
    float acc = bias[oc];
    int ih0 = oh * stride - pad;
    int iw0 = ow * stride - pad;
    for (int ic = 0; ic < IC; ++ic) {
        const float* ip = in + (size_t)ic * IH * IW;
        const float* wp = w + (((size_t)oc * IC + ic) * KH) * KW;
        for (int kh = 0; kh < KH; ++kh) {
            int ih = ih0 + kh;
            if (ih < 0 || ih >= IH) continue;
            const float* iprow = ip + ih * IW;
            const float* wprow = wp + kh * KW;
            for (int kw = 0; kw < KW; ++kw) {
                int iw = iw0 + kw;
                if (iw < 0 || iw >= IW) continue;
                acc += iprow[iw] * wprow[kw];
            }
        }
    }
    if (relu) acc = fmaxf(acc, 0.0f);
    out[idx] = acc;
}

// ---------------------------------------------------------------------------
// Local response norm across channels:  x / (k + (alpha/size)*sum x^2)^beta
// ---------------------------------------------------------------------------
__global__ void vpg_lrn_kernel(const float* __restrict__ in, float* __restrict__ out,
                               int C, int HW, float alpha_over_size, float beta, float k)
{
    int idx = blockIdx.x * blockDim.x + threadIdx.x;
    int total = C * HW;
    if (idx >= total) return;
    int p = idx % HW;
    int c = idx / HW;
    float s = 0.0f;
    for (int j = c - 2; j <= c + 2; ++j) {
        if (j >= 0 && j < C) {
            float v = in[(size_t)j * HW + p];
            s += v * v;
        }
    }
    out[idx] = in[idx] * powf(k + alpha_over_size * s, -beta);
}

// ---------------------------------------------------------------------------
// 3x3 stride-2 VALID max pool
// ---------------------------------------------------------------------------
__global__ void vpg_maxpool3s2_kernel(const float* __restrict__ in, float* __restrict__ out,
                                      int C, int IH, int IW, int OH, int OW)
{
    int idx = blockIdx.x * blockDim.x + threadIdx.x;
    int total = C * OH * OW;
    if (idx >= total) return;
    int ow = idx % OW;
    int oh = (idx / OW) % OH;
    int c = idx / (OW * OH);
    const float* ip = in + (size_t)c * IH * IW;
    int ih0 = oh * 2, iw0 = ow * 2;
    float m = -INFINITY;
    for (int r = 0; r < 3; ++r)
        for (int q = 0; q < 3; ++q)
            m = fmaxf(m, ip[(ih0 + r) * IW + iw0 + q]);
    out[idx] = m;
}

// ---------------------------------------------------------------------------
// Nearest-neighbor upsample: oi = (o * I) / O
// ---------------------------------------------------------------------------
__global__ void vpg_upsample_kernel(const float* __restrict__ in, float* __restrict__ out,
                                    int C, int IH, int IW, int OH, int OW)
{
    int idx = blockIdx.x * blockDim.x + threadIdx.x;
    int total = C * OH * OW;
    if (idx >= total) return;
    int ow = idx % OW;
    int oh = (idx / OW) % OH;
    int c = idx / (OW * OH);
    int ih = (oh * IH) / OH;
    int iw = (ow * IW) / OW;
    out[idx] = in[((size_t)c * IH + ih) * IW + iw];
}

// ---------------------------------------------------------------------------
// im2col for conv6 (6x6, pad 3) from x5 [384][14][19] into a B-fragment-
// ordered f16 panel (K=13824, Mpad=320; M=300 valid = 15x20 output grid).
// ---------------------------------------------------------------------------
__global__ void vpg_im2col6_kernel(const float* __restrict__ x5, half_t* __restrict__ B,
                                   int Mpad)
{
    int idx = blockIdx.x * blockDim.x + threadIdx.x;
    int total = 13824 * Mpad;
    if (idx >= total) return;
    int m = idx % Mpad;
    int k = idx / Mpad;
    half_t v = (half_t)0.0f;
    if (m < 300) {
        int ic = k / 36;
        int r = (k / 6) % 6;
        int c = k % 6;
        int oh = m / 20, ow = m % 20;
        int ih = oh - 3 + r;
        int iw = ow - 3 + c;
        if (ih >= 0 && ih < 14 && iw >= 0 && iw < 19)
            v = (half_t)x5[((size_t)ic * 14 + ih) * 19 + iw];
    }
    B[bfrag_idx(k, m, Mpad >> 4)] = v;
}

// ---------------------------------------------------------------------------
// Convert fp32 activation matrix [C][M] -> B-fragment-ordered f16 panel
// (C plays the role of K; zero-fill the m padding).
// ---------------------------------------------------------------------------
__global__ void vpg_f32_to_f16_frag_kernel(const float* __restrict__ in, half_t* __restrict__ out,
                                           int C, int M, int Mpad)
{
    int idx = blockIdx.x * blockDim.x + threadIdx.x;
    int total = C * Mpad;
    if (idx >= total) return;
    int m = idx % Mpad;
    int c = idx / Mpad;
    half_t v = (m < M) ? (half_t)in[(size_t)c * M + m] : (half_t)0.0f;
    out[bfrag_idx(c, m, Mpad >> 4)] = v;
}

// ---------------------------------------------------------------------------
// Pixel-shuffle "tile": in [OD*r*r][H][W] -> out [OD][H*r][W*r]
// ---------------------------------------------------------------------------
__global__ void vpg_tile_kernel(const float* __restrict__ in, float* __restrict__ out,
                                int OD, int r, int H, int W)
{
    int idx = blockIdx.x * blockDim.x + threadIdx.x;
    int OW2 = W * r;
    int OH2 = H * r;
    int total = OD * OH2 * OW2;
    if (idx >= total) return;
    int ow2 = idx % OW2;
    int oh2 = (idx / OW2) % OH2;
    int od = idx / (OW2 * OH2);
    int row = oh2 % r, col = ow2 % r;
    int h = oh2 / r, w = ow2 / r;
    int c = (od * r + row) * r + col;
    out[idx] = in[((size_t)c * H + h) * W + w];
}

// ---------------------------------------------------------------------------
// WMMA GEMM:  Out[n, m] = act( sum_k A[n,k]*B[k,m] + bias[n] )
//   A: weights  N x K fp32 row-major (cvt to f16 while staging through LDS)
//   B: f16 panel in B-fragment order (see bfrag_idx) -> fragments are direct
//      coalesced global_load_b128 pairs; no LDS, no barrier on the B path.
// Block = 256 threads (8 waves), block tile 64(N) x 64(M); each wave owns a
// 16(n) x 32(m) D tile (2 accumulators); K stepped by 64
// => 4 v_wmma_f32_16x16x32_f16 per thread per iteration.
//
// LDS holds only the A tile in fragment order (lane's 16 halfs contiguous):
//   lA[(kk*4 + s)*32 + lane][j], lane = hiL*16+mr holds row mr of subtile s,
//   slot j -> k: hiL*16+j
// Staged as global_load_b128 (float4) -> cvt -> ds_store_b64.
// No divergence before the WMMAs (EXEC all ones).
// ---------------------------------------------------------------------------
__global__ __launch_bounds__(256)
void vpg_gemm_wmma_kernel(const float* __restrict__ A,
                          const half_t* __restrict__ B,
                          const float* __restrict__ bias,
                          float* __restrict__ Out,
                          int N, int K, int M, int Mpad, int relu)
{
    __shared__ alignas(16) half_t lA[8 * 32 * 16];  // 8KB: [kk*4+s][lane][16]

    const int t    = threadIdx.x;
    const int wave = t >> 5;
    const int lane = t & 31;
    const int mr   = lane & 15;
    const int hi   = lane >> 4;
    const int Mp16 = Mpad >> 4;

    const int nTile = blockIdx.y * 64;
    const int mTile = blockIdx.x * 64;
    const int wn = (wave & 3) * 16;      // wave's n offset (A subtile)
    const int wm = (wave >> 2) * 32;     // wave's m offset (2 m-blocks)
    const int sA = wn >> 4;

    // ---- per-thread A staging coords: 4 groups of 4 contiguous k ----
    int aRow[4], aCol[4], aIdx[4];
    #pragma unroll
    for (int i = 0; i < 4; ++i) {
        int g  = t + i * 256;        // group of 4 elems of the 64x64 A tile
        int r  = g >> 4;             // tile row 0..63
        int c0 = (g & 15) << 2;      // tile col 0..60, step 4
        aRow[i] = r; aCol[i] = c0;
        int kk  = c0 >> 5;
        int cc  = c0 & 31;
        int hiA = (cc & 16) ? 16 : 0;
        int jb  = cc & 15;
        aIdx[i] = (((kk << 2) + (r >> 4)) * 32 + hiA + (r & 15)) * 16 + jb;
    }

    // B fragment bases: mb index = (mTile+wm)/16 + mc ; kb = k0/32 + kk
    const int mb0 = (mTile + wm) >> 4;

    v8f acc0 = {};
    v8f acc1 = {};

    for (int k0 = 0; k0 < K; k0 += 64) {
        const int kb0 = k0 >> 5;
        // -------- gather all globals into registers first (pipelined) -------
        v4f av[4];
        #pragma unroll
        for (int i = 0; i < 4; ++i)
            av[i] = *reinterpret_cast<const v4f*>(
                &A[(size_t)(nTile + aRow[i]) * K + (k0 + aCol[i])]);

        union Frag { v16h v; v8h q[2]; };
        Frag bf[2][2];   // [kk][mc]
        #pragma unroll
        for (int kk = 0; kk < 2; ++kk)
            #pragma unroll
            for (int mc = 0; mc < 2; ++mc) {
                const half_t* p = &B[((size_t)((kb0 + kk) * Mp16 + mb0 + mc) * 32 + lane) * 16];
                bf[kk][mc].q[0] = *reinterpret_cast<const v8h*>(p);
                bf[kk][mc].q[1] = *reinterpret_cast<const v8h*>(p + 8);
            }
        if (k0 + 64 < K)  // uniform branch: global_prefetch of next A panel
            __builtin_prefetch(&A[(size_t)(nTile + aRow[0]) * K + (k0 + 64 + aCol[0])], 0, 1);

        __syncthreads();   // previous iteration's A fragment reads are done

        // -------- store A to fragment-ordered LDS (cvt f32 -> f16) --------
        #pragma unroll
        for (int i = 0; i < 4; ++i) {
            v4h h;
            h[0] = (half_t)av[i][0];
            h[1] = (half_t)av[i][1];
            h[2] = (half_t)av[i][2];
            h[3] = (half_t)av[i][3];
            *reinterpret_cast<v4h*>(&lA[aIdx[i]]) = h;      // ds_store_b64
        }
        __syncthreads();

        // -------- A fragments: 2x ds_load_b128 each --------
        Frag a0, a1;
        const half_t* pa0 = &lA[((0 + sA) * 32 + lane) * 16];
        const half_t* pa1 = &lA[((4 + sA) * 32 + lane) * 16];
        a0.q[0] = *reinterpret_cast<const v8h*>(pa0);
        a0.q[1] = *reinterpret_cast<const v8h*>(pa0 + 8);
        a1.q[0] = *reinterpret_cast<const v8h*>(pa1);
        a1.q[1] = *reinterpret_cast<const v8h*>(pa1 + 8);

        acc0 = __builtin_amdgcn_wmma_f32_16x16x32_f16(
            false, a0.v, false, bf[0][0].v, (short)0, acc0, false, false);
        acc1 = __builtin_amdgcn_wmma_f32_16x16x32_f16(
            false, a0.v, false, bf[0][1].v, (short)0, acc1, false, false);
        acc0 = __builtin_amdgcn_wmma_f32_16x16x32_f16(
            false, a1.v, false, bf[1][0].v, (short)0, acc0, false, false);
        acc1 = __builtin_amdgcn_wmma_f32_16x16x32_f16(
            false, a1.v, false, bf[1][1].v, (short)0, acc1, false, false);
    }

    // ---- epilogue: D layout -> col = lane%16 (our m), row = v + 8*hi (our n)
    union { v8f v; float f[8]; } res[2];
    res[0].v = acc0;
    res[1].v = acc1;
    #pragma unroll
    for (int mc = 0; mc < 2; ++mc) {
        const int m = mTile + wm + mc * 16 + mr;
        if (m < M) {
            #pragma unroll
            for (int v = 0; v < 8; ++v) {
                int n = nTile + wn + v + 8 * hi;
                float val = res[mc].f[v] + bias[n];
                if (relu) val = fmaxf(val, 0.0f);
                Out[(size_t)n * M + m] = val;
            }
        }
    }
}

// ---------------------------------------------------------------------------
// Host-side orchestration
// ---------------------------------------------------------------------------
static inline dim3 grid1d(long long total) { return dim3((unsigned)((total + 255) / 256)); }

extern "C" void kernel_launch(void* const* d_in, const int* in_sizes, int n_in,
                              void* d_out, int out_size, void* d_ws, size_t ws_size,
                              hipStream_t stream)
{
    (void)in_sizes; (void)n_in; (void)out_size; (void)ws_size;

    const float* x   = (const float*)d_in[0];
    const float* w1  = (const float*)d_in[1];  const float* b1  = (const float*)d_in[2];
    const float* w2  = (const float*)d_in[3];  const float* b2  = (const float*)d_in[4];
    const float* w3  = (const float*)d_in[5];  const float* b3  = (const float*)d_in[6];
    const float* w4  = (const float*)d_in[7];  const float* b4  = (const float*)d_in[8];
    const float* w5  = (const float*)d_in[9];  const float* b5  = (const float*)d_in[10];
    const float* w6  = (const float*)d_in[11]; const float* b6  = (const float*)d_in[12];
    const float* w7a = (const float*)d_in[13]; const float* b7a = (const float*)d_in[14];
    const float* w8a = (const float*)d_in[15]; const float* b8a = (const float*)d_in[16];
    const float* w7b = (const float*)d_in[17]; const float* b7b = (const float*)d_in[18];
    const float* w8b = (const float*)d_in[19]; const float* b8b = (const float*)d_in[20];
    const float* w7c = (const float*)d_in[21]; const float* b7c = (const float*)d_in[22];
    const float* w8c = (const float*)d_in[23]; const float* b8c = (const float*)d_in[24];
    const float* w7d = (const float*)d_in[25]; const float* b7d = (const float*)d_in[26];
    const float* w8d = (const float*)d_in[27]; const float* b8d = (const float*)d_in[28];

    float* out = (float*)d_out;
    float* o_x6  = out;                                      // 4096*300
    float* o_x8a = out + 1228800;                            // 4*120*160
    float* o_x8b = out + 1228800 + 76800;                    // 2*120*160
    float* o_x8c = out + 1228800 + 76800 + 38400;            // 64*60*80
    float* o_x8d = out + 1228800 + 76800 + 38400 + 307200;   // 5*120*160

    char* ws = (char*)d_ws;
    size_t off = 0;
    auto alloc = [&](size_t bytes) -> void* {
        off = (off + 255) & ~(size_t)255;
        void* p = ws + off;
        off += bytes;
        return p;
    };

    const int M = 300, Mpad = 320;

    float* f1   = (float*)alloc((size_t)96 * 118 * 158 * 4);
    float* f1n  = (float*)alloc((size_t)96 * 118 * 158 * 4);
    float* f1p  = (float*)alloc((size_t)96 * 58 * 78 * 4);
    float* f1u  = (float*)alloc((size_t)96 * 59 * 79 * 4);
    float* f2   = (float*)alloc((size_t)256 * 59 * 79 * 4);
    float* f2n  = (float*)alloc((size_t)256 * 59 * 79 * 4);
    float* f2p  = (float*)alloc((size_t)256 * 29 * 39 * 4);
    float* f3   = (float*)alloc((size_t)384 * 29 * 39 * 4);
    float* f4   = (float*)alloc((size_t)384 * 29 * 39 * 4);
    float* f5c  = (float*)alloc((size_t)384 * 29 * 39 * 4);
    float* f5   = (float*)alloc((size_t)384 * 14 * 19 * 4);
    half_t* B6h = (half_t*)alloc((size_t)13824 * Mpad * 2);
    half_t* x6h = (half_t*)alloc((size_t)4096 * Mpad * 2);
    float* x7f  = (float*)alloc((size_t)4096 * M * 4);
    half_t* x7h = (half_t*)alloc((size_t)4096 * Mpad * 2);
    float* y8   = (float*)alloc((size_t)1024 * M * 4);

    // conv1 (3->96, 11x11, s4) + relu
    vpg_conv2d_kernel<<<grid1d((long long)96 * 118 * 158), 256, 0, stream>>>(
        x, w1, b1, f1, 3, 480, 640, 96, 118, 158, 11, 11, 4, 0, 1);
    vpg_lrn_kernel<<<grid1d((long long)96 * 118 * 158), 256, 0, stream>>>(
        f1, f1n, 96, 118 * 158, 0.0005f / 5.0f, 0.75f, 2.0f);
    vpg_maxpool3s2_kernel<<<grid1d((long long)96 * 58 * 78), 256, 0, stream>>>(
        f1n, f1p, 96, 118, 158, 58, 78);
    vpg_upsample_kernel<<<grid1d((long long)96 * 59 * 79), 256, 0, stream>>>(
        f1p, f1u, 96, 58, 78, 59, 79);

    // conv2 (96->256, 5x5, pad2) + relu, LRN, pool
    vpg_conv2d_kernel<<<grid1d((long long)256 * 59 * 79), 256, 0, stream>>>(
        f1u, w2, b2, f2, 96, 59, 79, 256, 59, 79, 5, 5, 1, 2, 1);
    vpg_lrn_kernel<<<grid1d((long long)256 * 59 * 79), 256, 0, stream>>>(
        f2, f2n, 256, 59 * 79, 0.0005f / 5.0f, 0.75f, 8.0f);
    vpg_maxpool3s2_kernel<<<grid1d((long long)256 * 29 * 39), 256, 0, stream>>>(
        f2n, f2p, 256, 59, 79, 29, 39);

    // conv3/4/5 (3x3 pad1) + relu, pool
    vpg_conv2d_kernel<<<grid1d((long long)384 * 29 * 39), 256, 0, stream>>>(
        f2p, w3, b3, f3, 256, 29, 39, 384, 29, 39, 3, 3, 1, 1, 1);
    vpg_conv2d_kernel<<<grid1d((long long)384 * 29 * 39), 256, 0, stream>>>(
        f3, w4, b4, f4, 384, 29, 39, 384, 29, 39, 3, 3, 1, 1, 1);
    vpg_conv2d_kernel<<<grid1d((long long)384 * 29 * 39), 256, 0, stream>>>(
        f4, w5, b5, f5c, 384, 29, 39, 384, 29, 39, 3, 3, 1, 1, 1);
    vpg_maxpool3s2_kernel<<<grid1d((long long)384 * 14 * 19), 256, 0, stream>>>(
        f5c, f5, 384, 29, 39, 14, 19);

    // conv6 as WMMA GEMM: N=4096, K=13824, M=300
    vpg_im2col6_kernel<<<grid1d((long long)13824 * Mpad), 256, 0, stream>>>(f5, B6h, Mpad);
    {
        dim3 g(Mpad / 64, 4096 / 64);
        vpg_gemm_wmma_kernel<<<g, 256, 0, stream>>>(w6, B6h, b6, o_x6, 4096, 13824, M, Mpad, 1);
    }
    vpg_f32_to_f16_frag_kernel<<<grid1d((long long)4096 * Mpad), 256, 0, stream>>>(
        o_x6, x6h, 4096, M, Mpad);

    // four heads: conv7 (4096->4096) + conv8 + pixel-shuffle tile
    struct Head {
        const float *w7, *b7, *w8, *b8;
        int oc8, r, od;
        float* o;
    } heads[4] = {
        { w7a, b7a, w8a, b8a,  256, 8,  4, o_x8a },
        { w7b, b7b, w8b, b8b,  128, 8,  2, o_x8b },
        { w7c, b7c, w8c, b8c, 1024, 4, 64, o_x8c },
        { w7d, b7d, w8d, b8d,  320, 8,  5, o_x8d },
    };
    for (int h = 0; h < 4; ++h) {
        const Head& H = heads[h];
        dim3 g7(Mpad / 64, 4096 / 64);
        vpg_gemm_wmma_kernel<<<g7, 256, 0, stream>>>(H.w7, x6h, H.b7, x7f, 4096, 4096, M, Mpad, 1);
        vpg_f32_to_f16_frag_kernel<<<grid1d((long long)4096 * Mpad), 256, 0, stream>>>(
            x7f, x7h, 4096, M, Mpad);
        dim3 g8(Mpad / 64, H.oc8 / 64);
        vpg_gemm_wmma_kernel<<<g8, 256, 0, stream>>>(H.w8, x7h, H.b8, y8, H.oc8, 4096, M, Mpad, 0);
        vpg_tile_kernel<<<grid1d((long long)H.od * 15 * H.r * 20 * H.r), 256, 0, stream>>>(
            y8, H.o, H.od, H.r, 15, 20);
    }
}
